// VectorQuantizer_9423158247847
// MI455X (gfx1250) — matrix operand order repaired and checked by reading, not compiled, not run on hardware
//
#include <hip/hip_runtime.h>

#define DIM 512
#define KCODES 8192
#define EMA_DECAY 0.99f
#define ONE_MINUS 0.01f
#define VQ_EPS 1e-5f
#define CCOST 0.25f

// Padded LDS row stride (in halfs): 520*2 = 1040 B -> 260 dwords -> lane nl's
// b128 window lands at bank (4*nl + 4*half) % 64: perfectly 2-way balanced.
#define BROW 520
#define NTILES (KCODES / 16)

typedef __attribute__((ext_vector_type(16))) _Float16 v16h;
typedef __attribute__((ext_vector_type(8)))  _Float16 v8h;
typedef __attribute__((ext_vector_type(8)))  float    v8f;
typedef __attribute__((ext_vector_type(4)))  float    v4f;

#if __has_builtin(__builtin_amdgcn_global_load_async_to_lds_b128)
#define USE_ASYNC_LDS 1
// Builtin signature (from hipcc diagnostic): takes generic pointers to
// "int __attribute__((vector_size(16)))".
typedef int v4i_async __attribute__((vector_size(16)));
#if __has_builtin(__builtin_amdgcn_s_wait_asynccnt)
#define ASYNC_WAIT() __builtin_amdgcn_s_wait_asynccnt(0)
#else
#define ASYNC_WAIT() asm volatile("s_wait_asynccnt 0x0" ::: "memory")
#endif
#else
#define USE_ASYNC_LDS 0
#define ASYNC_WAIT() ((void)0)
#endif

// ---------------------------------------------------------------------------
// Kernel 1: convert z (f32) -> f16, 8 elements per thread
// ---------------------------------------------------------------------------
__global__ void k_cvt_z_f16(const float* __restrict__ src,
                            _Float16* __restrict__ dst) {
    int t = blockIdx.x * blockDim.x + threadIdx.x;
    int i = t * 8;
    v4f a = *(const v4f*)(src + i);
    v4f b = *(const v4f*)(src + i + 4);
    v8h o;
#pragma unroll
    for (int j = 0; j < 4; ++j) {
        o[j]     = (_Float16)a[j];
        o[j + 4] = (_Float16)b[j];
    }
    *(v8h*)(dst + i) = o;
}

// ---------------------------------------------------------------------------
// Kernel 2: weight (f32) -> f16 + per-row squared L2 norm. One wave per row.
// ---------------------------------------------------------------------------
__global__ void k_prep_w(const float* __restrict__ w,
                         _Float16* __restrict__ w16,
                         float* __restrict__ wnorm) {
    int lane = threadIdx.x & 31;
    int wave = threadIdx.x >> 5;
    int row  = blockIdx.x * 8 + wave;
    const float* p = w + (size_t)row * DIM + lane * 16;
    float s = 0.0f;
    v8h o0, o1;
#pragma unroll
    for (int c = 0; c < 4; ++c) {
        v4f v = *(const v4f*)(p + c * 4);
#pragma unroll
        for (int j = 0; j < 4; ++j) {
            float x = v[j];
            s += x * x;
            if (c < 2) o0[c * 4 + j] = (_Float16)x;
            else       o1[(c - 2) * 4 + j] = (_Float16)x;
        }
    }
    _Float16* q = w16 + (size_t)row * DIM + lane * 16;
    *(v8h*)(q)     = o0;
    *(v8h*)(q + 8) = o1;
#pragma unroll
    for (int m = 16; m >= 1; m >>= 1) s += __shfl_xor(s, m, 32);
    if (lane == 0) wnorm[row] = s;
}

// ---------------------------------------------------------------------------
// Kernel 3: init EMA outputs with decayed old buffers; zero loss accumulator.
// ---------------------------------------------------------------------------
__global__ void k_init_ema(const float* __restrict__ ema_cs,
                           const float* __restrict__ ema_dw,
                           float* __restrict__ ncs_out,
                           float* __restrict__ dw_out,
                           float* __restrict__ lossAcc) {
    int t = blockIdx.x * blockDim.x + threadIdx.x;
    size_t i = (size_t)t * 4;
    v4f a = *(const v4f*)(ema_dw + i);
    a = a * EMA_DECAY;
    *(v4f*)(dw_out + i) = a;
    if (t < KCODES) ncs_out[t] = ema_cs[t] * EMA_DECAY;
    if (t == 0) lossAcc[0] = 0.0f;
}

// ---------------------------------------------------------------------------
// Stage one 16-code weight tile (16 rows x 512 f16) into LDS.
// 256 threads: thread = {row = tid>>4, c = tid&15}, copies 4 x 16B chunks.
// Async path: global -> LDS DMA, no VGPR involvement, tracked by ASYNCcnt.
// ---------------------------------------------------------------------------
__device__ __forceinline__ void stage_tile(const _Float16* __restrict__ w16,
                                           _Float16* sbase, int tile, int tid) {
    int row = tid >> 4;
    int c   = tid & 15;
    const _Float16* gp = w16 + (size_t)(tile * 16 + row) * DIM + c * 32;
    _Float16* lp = sbase + row * BROW + c * 32;
#if USE_ASYNC_LDS
#pragma unroll
    for (int j = 0; j < 4; ++j) {
        __builtin_amdgcn_global_load_async_to_lds_b128(
            (v4i_async*)(gp + j * 8), (v4i_async*)(lp + j * 8), 0, 0);
    }
#else
    v8h tmp[4];
#pragma unroll
    for (int j = 0; j < 4; ++j) tmp[j] = *(const v8h*)(gp + j * 8);
#pragma unroll
    for (int j = 0; j < 4; ++j) *(v8h*)(lp + j * 8) = tmp[j];
#endif
}

// ---------------------------------------------------------------------------
// Kernel 4: fused WMMA distance GEMM + argmin, LDS-staged codebook tiles.
// Block = 256 threads = 8 waves; wave w owns z rows [blk*128 + w*16, +16).
// A fragments (16x512 f16) live in 128 VGPRs for the whole code sweep; the
// shared B tile is fetched once per workgroup (async DMA), read 8x from LDS.
// argmin needs only wnorm[n] - 2*dot (|z|^2 is row-constant).
// ---------------------------------------------------------------------------
__global__ void __launch_bounds__(256)
k_argmin(const _Float16* __restrict__ z16,
         const _Float16* __restrict__ w16,
         const float* __restrict__ wnorm,
         int* __restrict__ idxI,
         float* __restrict__ idxF) {
    __shared__ _Float16 sbuf[2][16 * BROW];

    int tid  = threadIdx.x;
    int lane = tid & 31;
    int wave = tid >> 5;
    int lh   = lane >> 4;        // lane half
    int nl   = lane & 15;
    int rowBase = blockIdx.x * 128 + wave * 16;

    // Resident A fragments: z rows rowBase..rowBase+15, all K=512.
    // A 16-bit 16x32 layout: lane half h, VGPRs 0-3 = k h*8+0..7, 4-7 = 16+h*8+0..7
    const _Float16* zp = z16 + (size_t)(rowBase + nl) * DIM + lh * 8;
    v16h afrag[16];
#pragma unroll
    for (int kk = 0; kk < 16; ++kk) {
        const _Float16* p = zp + kk * 32;
        v8h lo = *(const v8h*)(p);
        v8h hi = *(const v8h*)(p + 16);
#pragma unroll
        for (int j = 0; j < 8; ++j) {
            afrag[kk][j]     = lo[j];
            afrag[kk][j + 8] = hi[j];
        }
    }

    float minval[8];
    int   minidx[8];
#pragma unroll
    for (int r = 0; r < 8; ++r) { minval[r] = 3.402823466e38f; minidx[r] = 0; }

    stage_tile(w16, &sbuf[0][0], 0, tid);

    for (int t = 0; t < NTILES; ++t) {
        int cur = t & 1;
        ASYNC_WAIT();          // my async stores for tile t have landed in LDS
        __syncthreads();       // tile t globally visible; buf[cur^1] free (its
                               // readers consumed their ds loads before arriving)
        if (t + 1 < NTILES) stage_tile(w16, &sbuf[cur ^ 1][0], t + 1, tid);

        // B 16-bit 32x16 layout: lane n = nl; half h holds k = h*8 + 0..7, 16 + h*8 + 0..7
        const _Float16* bb = &sbuf[cur][0] + nl * BROW + lh * 8;
        v8f acc = {};
#pragma unroll
        for (int kk = 0; kk < 16; ++kk) {
            v8h lo = *(const v8h*)(bb + kk * 32);
            v8h hi = *(const v8h*)(bb + kk * 32 + 16);
            v16h b;
#pragma unroll
            for (int j = 0; j < 8; ++j) { b[j] = lo[j]; b[j + 8] = hi[j]; }
            acc = __builtin_amdgcn_wmma_f32_16x16x32_f16(
                false, afrag[kk], false, b, (short)0, acc, false, false);
        }
        float wn = wnorm[t * 16 + nl];
#pragma unroll
        for (int r = 0; r < 8; ++r) {
            float dd = __builtin_fmaf(-2.0f, acc[r], wn);
            if (dd < minval[r]) { minval[r] = dd; minidx[r] = t * 16 + nl; }
        }
    }

    // Butterfly argmin across the 16 lanes sharing each output row.
#pragma unroll
    for (int r = 0; r < 8; ++r) {
        float v = minval[r];
        int   i = minidx[r];
#pragma unroll
        for (int m = 8; m >= 1; m >>= 1) {
            float ov = __shfl_xor(v, m, 32);
            int   oi = __shfl_xor(i, m, 32);
            if (ov < v || (ov == v && oi < i)) { v = ov; i = oi; }
        }
        minval[r] = v;
        minidx[r] = i;
    }
    if (nl == 0) {
        int base = rowBase + lh * 8;   // lane0 -> rows 0..7, lane16 -> 8..15
#pragma unroll
        for (int r = 0; r < 8; ++r) {
            idxI[base + r] = minidx[r];
            idxF[base + r] = (float)minidx[r];
        }
    }
}

// ---------------------------------------------------------------------------
// Kernel 5: gather codebook rows -> quantized, commitment loss partial,
// scatter-add EMA updates. One block (256 thr) per z row, 2 elems/thread.
// ---------------------------------------------------------------------------
__global__ void __launch_bounds__(256)
k_gather_scatter(const float* __restrict__ z,
                 const float* __restrict__ weight,
                 const int* __restrict__ idxI,
                 float* __restrict__ quantized,
                 float* __restrict__ dw_out,
                 float* __restrict__ ncs_out,
                 float* __restrict__ lossAcc) {
    __shared__ float sred[256];
    int b = blockIdx.x;
    int t = threadIdx.x;
    int e = idxI[b];
    float ls = 0.0f;
#pragma unroll
    for (int j = 0; j < 2; ++j) {
        int d = t + j * 256;
        float q  = weight[(size_t)e * DIM + d];
        float zv = z[(size_t)b * DIM + d];
        quantized[(size_t)b * DIM + d] = q;
        float df = zv - q;
        ls += df * df;
        atomicAdd(dw_out + (size_t)e * DIM + d, ONE_MINUS * zv);
    }
    sred[t] = ls;
    __syncthreads();
#pragma unroll
    for (int s = 128; s > 0; s >>= 1) {
        if (t < s) sred[t] += sred[t + s];
        __syncthreads();
    }
    if (t == 0) {
        atomicAdd(lossAcc, sred[0]);
        atomicAdd(ncs_out + e, ONE_MINUS);
    }
}

// ---------------------------------------------------------------------------
// Kernel 6: n = sum(new_cluster_size); finalize commitment loss scalar.
// ---------------------------------------------------------------------------
__global__ void k_finalize_scalars(const float* __restrict__ ncs,
                                   const float* __restrict__ lossAcc,
                                   float* __restrict__ nSum,
                                   float* __restrict__ lossOut,
                                   int Btotal) {
    __shared__ float sred[256];
    int t = threadIdx.x;
    float s = 0.0f;
    for (int i = t; i < KCODES; i += 256) s += ncs[i];
    sred[t] = s;
    __syncthreads();
#pragma unroll
    for (int ss = 128; ss > 0; ss >>= 1) {
        if (t < ss) sred[t] += sred[t + ss];
        __syncthreads();
    }
    if (t == 0) {
        nSum[0] = sred[0];
        lossOut[0] = lossAcc[0] * (CCOST / ((float)Btotal * (float)DIM));
    }
}

// ---------------------------------------------------------------------------
// Kernel 7: new_weight = new_ema_dw / laplace-smoothed cluster size.
// ---------------------------------------------------------------------------
__global__ void k_new_weight(const float* __restrict__ dw_out,
                             const float* __restrict__ ncs,
                             const float* __restrict__ nSum,
                             float* __restrict__ wOut) {
    int t = blockIdx.x * blockDim.x + threadIdx.x;
    size_t i = (size_t)t * 4;
    int k = (int)(i >> 9);   // / DIM
    float n = nSum[0];
    float cs = (ncs[k] + VQ_EPS) / (n + (float)KCODES * VQ_EPS) * n;
    v4f a = *(const v4f*)(dw_out + i);
    a = a / cs;
    *(v4f*)(wOut + i) = a;
}

// ---------------------------------------------------------------------------
extern "C" void kernel_launch(void* const* d_in, const int* in_sizes, int n_in,
                              void* d_out, int out_size, void* d_ws, size_t ws_size,
                              hipStream_t stream) {
    const float* z       = (const float*)d_in[0];
    const float* weight  = (const float*)d_in[1];
    const float* ema_cs  = (const float*)d_in[2];
    const float* ema_dw  = (const float*)d_in[3];

    int B = in_sizes[0] / DIM;   // 32768

    // d_out layout: quantized (B*D) | indices (B) | loss (1) |
    //               new_weight (K*D) | new_cluster_size (K) | new_ema_dw (K*D)
    float* out       = (float*)d_out;
    float* quantized = out;
    float* idxF      = out + (size_t)B * DIM;
    float* lossOut   = idxF + B;
    float* wOut      = lossOut + 1;
    float* ncsOut    = wOut + (size_t)KCODES * DIM;
    float* dwOut     = ncsOut + KCODES;

    // workspace layout
    char* ws = (char*)d_ws;
    _Float16* z16 = (_Float16*)ws;                               // B*D*2
    _Float16* w16 = (_Float16*)(ws + (size_t)B * DIM * 2);       // K*D*2
    float* wnorm  = (float*)(ws + (size_t)B * DIM * 2 + (size_t)KCODES * DIM * 2);
    int*   idxI   = (int*)((char*)wnorm + (size_t)KCODES * 4);
    float* lossAcc = (float*)((char*)idxI + (size_t)B * 4);
    float* nSum    = lossAcc + 1;

    k_cvt_z_f16<<<(B * DIM) / 2048, 256, 0, stream>>>(z, z16);
    k_prep_w<<<KCODES / 8, 256, 0, stream>>>(weight, w16, wnorm);
    k_init_ema<<<(KCODES * DIM / 4) / 256, 256, 0, stream>>>(ema_cs, ema_dw,
                                                             ncsOut, dwOut, lossAcc);
    k_argmin<<<B / 128, 256, 0, stream>>>(z16, w16, wnorm, idxI, idxF);
    k_gather_scatter<<<B, 256, 0, stream>>>(z, weight, idxI, quantized,
                                            dwOut, ncsOut, lossAcc);
    k_finalize_scalars<<<1, 256, 0, stream>>>(ncsOut, lossAcc, nSum, lossOut, B);
    k_new_weight<<<(KCODES * DIM / 4) / 256, 256, 0, stream>>>(dwOut, ncsOut,
                                                               nSum, wOut);
}
